// SubgraphNet_55070070669490
// MI455X (gfx1250) — compile-verified, matrix operand order checked by reading, not run on hardware
//
#include <hip/hip_runtime.h>
#include <hip/hip_fp16.h>

typedef __attribute__((ext_vector_type(16))) _Float16 v16h;
typedef __attribute__((ext_vector_type(8)))  float    v8f;

#define LDS_FENCE() asm volatile("s_wait_dscnt 0" ::: "memory")

#define HID   128
#define HEADS 16
#define DH    8

// ---------------------------------------------------------------------------
// Float atomic max via monotonic integer-bits trick (init must be -inf).
// ---------------------------------------------------------------------------
__device__ __forceinline__ void atomicMaxF(float* addr, float val) {
    if (val >= 0.0f) atomicMax((int*)addr, __float_as_int(val));
    else             atomicMin((unsigned int*)addr, __float_as_uint(val));
}

// ---------------------------------------------------------------------------
// WMMA fragment helpers.
// A (16x32 f16), per ISA layout: lanes 0-15 row=lane hold K {0..7,16..23},
// lanes 16-31 row=lane-16 hold K {8..15,24..31}. Two 16B LDS loads.
// ---------------------------------------------------------------------------
__device__ __forceinline__ v16h load_afrag(const _Float16* lds, int kpad,
                                           int row, int k0, int lane) {
    int koff = (lane >= 16) ? 8 : 0;
    const _Float16* p = lds + row * kpad + k0 + koff;
    union { v16h v; uint4 u[2]; } U;
    U.u[0] = *(const uint4*)(p);
    U.u[1] = *(const uint4*)(p + 16);
    return U.v;
}

// B fragments pre-packed in global memory: [nt][kt][lane][16 halves]
__device__ __forceinline__ v16h load_bfrag(const _Float16* wpk, int nt, int kt,
                                           int KT, int lane) {
    const _Float16* p = wpk + (size_t)(((nt * KT + kt) * 32) + lane) * 16;
    union { v16h v; uint4 u[2]; } U;
    U.u[0] = ((const uint4*)p)[0];
    U.u[1] = ((const uint4*)p)[1];
    return U.v;
}

// 16xK @ Kx128 -> 16x128 in 8 v8f accumulators (node-side kernels).
template <int KT>
__device__ __forceinline__ void gemm16(const _Float16* aLds, int kpad,
                                       const _Float16* wpk, v8f* acc, int lane) {
#pragma unroll
    for (int kt = 0; kt < KT; ++kt) {
        v16h a = load_afrag(aLds, kpad, lane & 15, kt * 32, lane);
#pragma unroll
        for (int nt = 0; nt < 8; ++nt) {
            v16h b  = load_bfrag(wpk, nt, kt, KT, lane);
            acc[nt] = __builtin_amdgcn_wmma_f32_16x16x32_f16(
                false, a, false, b, (short)0, acc[nt], false, false);
        }
    }
}

// 32xK @ Kx128 -> 32x128: two A fragments share every B fragment
// (16 WMMA per 8 B-fragment loads -> half the VMEM per WMMA).
template <int KT>
__device__ __forceinline__ void gemm32(const _Float16* aLds, int kpad,
                                       const _Float16* wpk,
                                       v8f* accL, v8f* accH, int lane) {
    int arow = lane & 15;
#pragma unroll
    for (int kt = 0; kt < KT; ++kt) {
        v16h a0 = load_afrag(aLds, kpad, arow,      kt * 32, lane);
        v16h a1 = load_afrag(aLds, kpad, 16 + arow, kt * 32, lane);
#pragma unroll
        for (int nt = 0; nt < 8; ++nt) {
            v16h b = load_bfrag(wpk, nt, kt, KT, lane);
            accL[nt] = __builtin_amdgcn_wmma_f32_16x16x32_f16(
                false, a0, false, b, (short)0, accL[nt], false, false);
            accH[nt] = __builtin_amdgcn_wmma_f32_16x16x32_f16(
                false, a1, false, b, (short)0, accH[nt], false, false);
        }
    }
}

// C layout: vgpr r, lanes 0-15 -> (row r, col lane); lanes 16-31 -> (row r+8,
// col lane-16). Dump (+bias) to LDS row-major f32.
__device__ __forceinline__ void dump_c(const v8f* acc, float* h1,
                                       const float* bias, int lane) {
    int rowOff = (lane >= 16) ? 8 : 0;
    int col    = lane & 15;
#pragma unroll
    for (int nt = 0; nt < 8; ++nt) {
        float b = bias[nt * 16 + col];
#pragma unroll
        for (int r = 0; r < 8; ++r)
            h1[(r + rowOff) * HID + nt * 16 + col] = acc[nt][r] + b;
    }
    LDS_FENCE();
}

__device__ __forceinline__ void dump_c32(const v8f* accL, const v8f* accH,
                                         float* h1, const float* bias, int lane) {
    int rowOff = (lane >= 16) ? 8 : 0;
    int col    = lane & 15;
#pragma unroll
    for (int nt = 0; nt < 8; ++nt) {
        float b = bias[nt * 16 + col];
#pragma unroll
        for (int r = 0; r < 8; ++r) {
            h1[(r + rowOff) * HID + nt * 16 + col]        = accL[nt][r] + b;
            h1[(16 + r + rowOff) * HID + nt * 16 + col]   = accH[nt][r] + b;
        }
    }
    LDS_FENCE();
}

// LayerNorm + ReLU on 16x128 f32 tile -> f16 A-tile (node-side).
__device__ __forceinline__ void ln_relu(const float* h1, _Float16* a2,
                                        float* stats, const float* g,
                                        const float* beta, int lane) {
    int row = lane & 15;
    int c0  = (lane >> 4) * 64;
    float s = 0.f, sq = 0.f;
    for (int j = 0; j < 64; ++j) {
        int   c = c0 + ((j + row * 4) & 63);
        float x = h1[row * HID + c];
        s += x; sq += x * x;
    }
    s  += __shfl_xor(s, 16, 32);
    sq += __shfl_xor(sq, 16, 32);
    float mu  = s * (1.0f / 128.0f);
    float var = sq * (1.0f / 128.0f) - mu * mu;
    float rs  = rsqrtf(var + 1e-5f);
    if (lane < 16) { stats[row * 2] = mu; stats[row * 2 + 1] = rs; }
    LDS_FENCE();
    for (int i = 0; i < 64; ++i) {
        int   idx = i * 32 + lane;
        int   r = idx >> 7, c = idx & 127;
        float y = (h1[idx] - stats[r * 2]) * stats[r * 2 + 1] * g[c] + beta[c];
        a2[idx] = (_Float16)fmaxf(y, 0.0f);
    }
    LDS_FENCE();
}

// LayerNorm + ReLU on 32x128 f32 tile: one full row per lane for stats.
__device__ __forceinline__ void ln_relu32(const float* h1, _Float16* a2,
                                          float* stats, const float* g,
                                          const float* beta, int lane) {
    float s = 0.f, sq = 0.f;
    for (int j = 0; j < 128; ++j) {
        int   c = (j + lane * 4) & 127;   // stagger LDS banks across rows
        float x = h1[lane * HID + c];
        s += x; sq += x * x;
    }
    float mu  = s * (1.0f / 128.0f);
    float var = sq * (1.0f / 128.0f) - mu * mu;
    float rs  = rsqrtf(var + 1e-5f);
    stats[lane * 2]     = mu;
    stats[lane * 2 + 1] = rs;
    LDS_FENCE();
    for (int i = 0; i < 128; ++i) {
        int   idx = i * 32 + lane;
        int   r = idx >> 7, c = idx & 127;
        float y = (h1[idx] - stats[r * 2]) * stats[r * 2 + 1] * g[c] + beta[c];
        a2[idx] = (_Float16)fmaxf(y, 0.0f);
    }
    LDS_FENCE();
}

// ---------------------------------------------------------------------------
// Weight repack: f32 row-major [K][128] -> f16 B-fragments [nt][kt][lane][16].
// ---------------------------------------------------------------------------
__global__ void pack_weights(const float* __restrict__ W, _Float16* __restrict__ dst,
                             int K, int KT) {
    int tid   = blockIdx.x * blockDim.x + threadIdx.x;
    int total = 8 * KT * 32 * 16;
    if (tid >= total) return;
    int hh   = tid & 15;
    int lane = (tid >> 4) & 31;
    int kt   = (tid >> 9) % KT;
    int nt   = tid / (512 * KT);
    int col  = nt * 16 + (lane & 15);
    int koff = ((hh < 8) ? 0 : 16) + ((lane >= 16) ? 8 : 0) + (hh & 7);
    int kk   = kt * 32 + koff;
    dst[tid] = (_Float16)((kk < K) ? W[kk * HID + col] : 0.0f);
}

__global__ void init_kernel(float* segmax, float* segsum, float* agg, int N) {
    int t = blockIdx.x * blockDim.x + threadIdx.x;
    int a = N * HEADS;
    if (t < a)                    segmax[t]       = -INFINITY;
    else if (t < 2 * a)           segsum[t - a]   = 0.0f;
    else if (t < 2 * a + N * HID) agg[t - 2 * a]  = 0.0f;
}

// ---------------------------------------------------------------------------
// q = MLP_q(h), output f16 [N][128].  (M=16, two waves per block)
// ---------------------------------------------------------------------------
__global__ __launch_bounds__(64) void mlp_q_kernel(
    const float* __restrict__ h, const _Float16* W1p, const float* b1,
    const float* g, const float* beta, const _Float16* W2p, const float* b2,
    _Float16* __restrict__ qOut, int nTiles) {
    __shared__ _Float16 aIn[2][16 * HID];
    __shared__ float    h1[2][16 * HID];
    __shared__ _Float16 a2[2][16 * HID];
    __shared__ float    stats[2][32];
    int wave = threadIdx.x >> 5, lane = threadIdx.x & 31;
    int tile = blockIdx.x * 2 + wave;
    if (tile >= nTiles) return;
    int base = tile * 16;

    for (int i = 0; i < 64; ++i) {
        int idx = i * 32 + lane;
        aIn[wave][idx] = (_Float16)h[(size_t)base * HID + idx];
    }
    LDS_FENCE();
    v8f acc[8] = {};
    gemm16<4>(aIn[wave], HID, W1p, acc, lane);
    dump_c(acc, h1[wave], b1, lane);
    ln_relu(h1[wave], a2[wave], stats[wave], g, beta, lane);
    v8f acc2[8] = {};
    gemm16<4>(a2[wave], HID, W2p, acc2, lane);

    int rowOff = (lane >= 16) ? 8 : 0, col = lane & 15;
#pragma unroll
    for (int nt = 0; nt < 8; ++nt) {
        float b = b2[nt * 16 + col];
#pragma unroll
        for (int r = 0; r < 8; ++r)
            qOut[(size_t)(base + r + rowOff) * HID + nt * 16 + col] =
                (_Float16)(acc2[nt][r] + b);
    }
}

// ---------------------------------------------------------------------------
// Build kv_input tile [32][160] f16 (h[src] | edge_attr | zero pad).
// ---------------------------------------------------------------------------
__device__ __forceinline__ void build_kv32(const float* h, const float* edge_attr,
                                           const int* sidx, long long eBase,
                                           long long E, _Float16* kvA, int lane) {
    union { _Float16 x[4]; uint2 u; } P;
    // 32 rows x 128 f32 gathered as float4 -> packed half4 LDS writes
    for (int i = 0; i < 32; ++i) {
        int c = i * 32 + lane;            // 0..1023 chunks of 4
        int r = c >> 5, q4 = c & 31;
        float4 f = ((const float4*)(h + (size_t)sidx[r] * HID))[q4];
        P.x[0] = (_Float16)f.x; P.x[1] = (_Float16)f.y;
        P.x[2] = (_Float16)f.z; P.x[3] = (_Float16)f.w;
        *(uint2*)&kvA[r * 160 + q4 * 4] = P.u;
    }
    // 32 rows x 16 edge features
    for (int i = 0; i < 4; ++i) {
        int c = i * 32 + lane;            // 0..127 chunks of 4
        int r = c >> 2, q4 = c & 3;
        long long er = eBase + r; if (er >= E) er = E - 1;
        float4 f = ((const float4*)(edge_attr + er * 16))[q4];
        P.x[0] = (_Float16)f.x; P.x[1] = (_Float16)f.y;
        P.x[2] = (_Float16)f.z; P.x[3] = (_Float16)f.w;
        *(uint2*)&kvA[r * 160 + 128 + q4 * 4] = P.u;
    }
    // zero pad cols 144..159
    for (int i = 0; i < 4; ++i) {
        int c = i * 32 + lane;
        int r = c >> 2, q4 = c & 3;
        *(uint2*)&kvA[r * 160 + 144 + q4 * 4] = make_uint2(0u, 0u);
    }
    LDS_FENCE();
}

// ---------------------------------------------------------------------------
// Edge pass 1: k-MLP (M=32) + attention logits + segmented max.
// One wave per block; ~35 KB LDS.
// ---------------------------------------------------------------------------
__global__ __launch_bounds__(32) void edge_k_logits_kernel(
    const float* __restrict__ h, const float* __restrict__ edge_attr,
    const int* __restrict__ edge_index, const _Float16* W1p, const float* b1,
    const float* g, const float* beta, const _Float16* W2p, const float* b2,
    const _Float16* __restrict__ q, float* __restrict__ logitsBuf,
    float* __restrict__ segmax, int nTiles, long long E) {
    __shared__ _Float16 kvA[32 * 160];     // reused as qT after GEMM1
    __shared__ float    h1[32 * HID];
    __shared__ _Float16 a2[32 * HID];
    __shared__ float    stats[64];
    __shared__ int      sidx[32], didx[32];
    int lane = threadIdx.x;
    int tile = blockIdx.x;
    if (tile >= nTiles) return;
    long long eBase = (long long)tile * 32;

    {
        long long ee = eBase + lane; if (ee >= E) ee = E - 1;
        sidx[lane] = edge_index[ee];
        didx[lane] = edge_index[E + ee];
    }
    LDS_FENCE();
    build_kv32(h, edge_attr, sidx, eBase, E, kvA, lane);

    v8f accL[8] = {}, accH[8] = {};
    gemm32<5>(kvA, 160, W1p, accL, accH, lane);
    dump_c32(accL, accH, h1, b1, lane);
    ln_relu32(h1, a2, stats, g, beta, lane);
    v8f acc2L[8] = {}, acc2H[8] = {};
    gemm32<4>(a2, HID, W2p, acc2L, acc2H, lane);
    dump_c32(acc2L, acc2H, h1, b2, lane);     // k tile f32 (incl. b2)

    // gather q[dst] (f16, 256B rows) into recycled kvA region
    _Float16* qT = kvA;
    for (int i = 0; i < 16; ++i) {
        int c = i * 32 + lane;                // 0..511 uint4 chunks
        int r = c >> 4, o = c & 15;
        ((uint4*)&qT[r * HID])[o] =
            ((const uint4*)(q + (size_t)didx[r] * HID))[o];
    }
    LDS_FENCE();

    // logits[e, head] = (q[dst] . k) / sqrt(8)
    for (int i = 0; i < 16; ++i) {
        int p = i * 32 + lane;                // 0..511
        int m = p >> 4, hd = p & 15;
        float d = 0.f;
#pragma unroll
        for (int j = 0; j < DH; ++j)
            d += (float)qT[m * HID + hd * DH + j] * h1[m * HID + hd * DH + j];
        float lg = d * 0.35355339059f;
        if (eBase + m < E) {
            logitsBuf[(eBase + m) * HEADS + hd] = lg;
            atomicMaxF(&segmax[(size_t)didx[m] * HEADS + hd], lg);
        }
    }
}

// ---------------------------------------------------------------------------
// Edge pass 2: ex = exp(logit - segmax[dst]); segsum += ex.  (in-place)
// ---------------------------------------------------------------------------
__global__ void softmax_edge_kernel(const int* __restrict__ edge_index,
                                    float* __restrict__ exBuf,
                                    const float* __restrict__ segmax,
                                    float* __restrict__ segsum, long long E) {
    long long t = (long long)blockIdx.x * blockDim.x + threadIdx.x;
    if (t >= E * HEADS) return;
    long long e = t >> 4;
    int hd = (int)(t & 15);
    int d  = edge_index[E + e];
    float ex = __expf(exBuf[t] - segmax[(size_t)d * HEADS + hd]);
    exBuf[t] = ex;
    atomicAdd(&segsum[(size_t)d * HEADS + hd], ex);
}

// ---------------------------------------------------------------------------
// Edge pass 3: v-MLP (M=32, recomputed) + alpha * v scattered into agg.
// ---------------------------------------------------------------------------
__global__ __launch_bounds__(32) void edge_v_agg_kernel(
    const float* __restrict__ h, const float* __restrict__ edge_attr,
    const int* __restrict__ edge_index, const _Float16* W1p, const float* b1,
    const float* g, const float* beta, const _Float16* W2p, const float* b2,
    const float* __restrict__ exBuf, const float* __restrict__ segsum,
    float* __restrict__ agg, int nTiles, long long E) {
    __shared__ _Float16 kvA[32 * 160];
    __shared__ float    h1[32 * HID];
    __shared__ _Float16 a2[32 * HID];
    __shared__ float    stats[64];
    __shared__ float    alph[32 * 16];
    __shared__ int      sidx[32], didx[32];
    int lane = threadIdx.x;
    int tile = blockIdx.x;
    if (tile >= nTiles) return;
    long long eBase = (long long)tile * 32;

    {
        long long ee = eBase + lane; if (ee >= E) ee = E - 1;
        sidx[lane] = edge_index[ee];
        didx[lane] = edge_index[E + ee];
    }
    LDS_FENCE();
    build_kv32(h, edge_attr, sidx, eBase, E, kvA, lane);

    v8f accL[8] = {}, accH[8] = {};
    gemm32<5>(kvA, 160, W1p, accL, accH, lane);
    dump_c32(accL, accH, h1, b1, lane);
    ln_relu32(h1, a2, stats, g, beta, lane);
    v8f acc2L[8] = {}, acc2H[8] = {};
    gemm32<4>(a2, HID, W2p, acc2L, acc2H, lane);
    dump_c32(acc2L, acc2H, h1, b2, lane);     // v tile f32

    for (int i = 0; i < 16; ++i) {
        int p = i * 32 + lane;
        int m = p >> 4, hd = p & 15;
        long long em = eBase + m; if (em >= E) em = E - 1;
        alph[m * 16 + hd] = exBuf[em * HEADS + hd] /
                            segsum[(size_t)didx[m] * HEADS + hd];
    }
    LDS_FENCE();
    for (int i = 0; i < 128; ++i) {
        int idx = i * 32 + lane;
        int r = idx >> 7, c = idx & 127;
        if (eBase + r < E)
            atomicAdd(&agg[(size_t)didx[r] * HID + c],
                      alph[r * 16 + (c >> 3)] * h1[idx]);
    }
}

// ---------------------------------------------------------------------------
// node_out MLP on concat(agg, h) + residual.  (M=16, two waves per block)
// ---------------------------------------------------------------------------
__global__ __launch_bounds__(64) void node_out_kernel(
    const float* __restrict__ h, const float* __restrict__ agg,
    const _Float16* W1p, const float* b1, const float* g, const float* beta,
    const _Float16* W2p, const float* b2, float* __restrict__ out, int nTiles) {
    __shared__ _Float16 aIn[2][16 * 256];
    __shared__ float    h1[2][16 * HID];
    __shared__ _Float16 a2[2][16 * HID];
    __shared__ float    stats[2][32];
    int wave = threadIdx.x >> 5, lane = threadIdx.x & 31;
    int tile = blockIdx.x * 2 + wave;
    if (tile >= nTiles) return;
    int base = tile * 16;

    for (int i = 0; i < 64; ++i) {
        int idx = i * 32 + lane;
        int r = idx >> 7, c = idx & 127;
        aIn[wave][r * 256 + c]       = (_Float16)agg[(size_t)(base + r) * HID + c];
        aIn[wave][r * 256 + 128 + c] = (_Float16)h[(size_t)(base + r) * HID + c];
    }
    LDS_FENCE();
    v8f acc[8] = {};
    gemm16<8>(aIn[wave], 256, W1p, acc, lane);
    dump_c(acc, h1[wave], b1, lane);
    ln_relu(h1[wave], a2[wave], stats[wave], g, beta, lane);
    v8f acc2[8] = {};
    gemm16<4>(a2[wave], HID, W2p, acc2, lane);

    int rowOff = (lane >= 16) ? 8 : 0, col = lane & 15;
#pragma unroll
    for (int nt = 0; nt < 8; ++nt) {
        float b = b2[nt * 16 + col];
#pragma unroll
        for (int r = 0; r < 8; ++r) {
            size_t o = (size_t)(base + r + rowOff) * HID + nt * 16 + col;
            out[o] = acc2[nt][r] + b + h[o];
        }
    }
}

// ---------------------------------------------------------------------------
extern "C" void kernel_launch(void* const* d_in, const int* in_sizes, int n_in,
                              void* d_out, int out_size, void* d_ws, size_t ws_size,
                              hipStream_t stream) {
    const float* h         = (const float*)d_in[0];
    const float* edge_attr = (const float*)d_in[1];
    const int*   edge_idx  = (const int*)d_in[2];
    // params flattened in dict order: hk, hv, hq, node_out; each W1,b1,g,beta,W2,b2
    const float* hkW1 = (const float*)d_in[3];
    const float* hkb1 = (const float*)d_in[4];
    const float* hkg  = (const float*)d_in[5];
    const float* hkbe = (const float*)d_in[6];
    const float* hkW2 = (const float*)d_in[7];
    const float* hkb2 = (const float*)d_in[8];
    const float* hvW1 = (const float*)d_in[9];
    const float* hvb1 = (const float*)d_in[10];
    const float* hvg  = (const float*)d_in[11];
    const float* hvbe = (const float*)d_in[12];
    const float* hvW2 = (const float*)d_in[13];
    const float* hvb2 = (const float*)d_in[14];
    const float* hqW1 = (const float*)d_in[15];
    const float* hqb1 = (const float*)d_in[16];
    const float* hqg  = (const float*)d_in[17];
    const float* hqbe = (const float*)d_in[18];
    const float* hqW2 = (const float*)d_in[19];
    const float* hqb2 = (const float*)d_in[20];
    const float* noW1 = (const float*)d_in[21];
    const float* nob1 = (const float*)d_in[22];
    const float* nog  = (const float*)d_in[23];
    const float* nobe = (const float*)d_in[24];
    const float* noW2 = (const float*)d_in[25];
    const float* nob2 = (const float*)d_in[26];

    const int       N = in_sizes[0] / HID;        // 50000
    const long long E = in_sizes[1] / 16;         // 1600000

    // workspace carve (256B aligned)
    char*  ws  = (char*)d_ws;
    size_t off = 0;
    auto carve = [&](size_t bytes) {
        void* p = ws + off;
        off += (bytes + 255) & ~(size_t)255;
        return p;
    };
    _Float16* qbuf   = (_Float16*)carve((size_t)N * HID * 2);
    float*    logits = (float*)carve((size_t)E * HEADS * 4);
    float*    segmax = (float*)carve((size_t)N * HEADS * 4);
    float*    segsum = (float*)carve((size_t)N * HEADS * 4);
    float*    agg    = (float*)carve((size_t)N * HID * 4);
    _Float16* hkW1p  = (_Float16*)carve(5 * 8192);   // KT=5 (K=144 pad 160)
    _Float16* hkW2p  = (_Float16*)carve(4 * 8192);
    _Float16* hvW1p  = (_Float16*)carve(5 * 8192);
    _Float16* hvW2p  = (_Float16*)carve(4 * 8192);
    _Float16* hqW1p  = (_Float16*)carve(4 * 8192);
    _Float16* hqW2p  = (_Float16*)carve(4 * 8192);
    _Float16* noW1p  = (_Float16*)carve(8 * 8192);   // KT=8 (K=256)
    _Float16* noW2p  = (_Float16*)carve(4 * 8192);
    (void)ws_size;

    // 0) weight repack + init
    auto pack = [&](const float* W, _Float16* dst, int K, int KT) {
        int total = 4096 * KT;
        pack_weights<<<(total + 255) / 256, 256, 0, stream>>>(W, dst, K, KT);
    };
    pack(hkW1, hkW1p, 144, 5); pack(hkW2, hkW2p, 128, 4);
    pack(hvW1, hvW1p, 144, 5); pack(hvW2, hvW2p, 128, 4);
    pack(hqW1, hqW1p, 128, 4); pack(hqW2, hqW2p, 128, 4);
    pack(noW1, noW1p, 256, 8); pack(noW2, noW2p, 128, 4);

    int initTot = N * HEADS * 2 + N * HID;
    init_kernel<<<(initTot + 255) / 256, 256, 0, stream>>>(segmax, segsum, agg, N);

    // 1) q MLP
    int nTilesN = N / 16;
    mlp_q_kernel<<<(nTilesN + 1) / 2, 64, 0, stream>>>(
        h, hqW1p, hqb1, hqg, hqbe, hqW2p, hqb2, qbuf, nTilesN);

    // 2) edge k-MLP + logits + segmax   (M=32 tiles, one wave per block)
    int nT32 = (int)((E + 31) / 32);
    edge_k_logits_kernel<<<nT32, 32, 0, stream>>>(
        h, edge_attr, edge_idx, hkW1p, hkb1, hkg, hkbe, hkW2p, hkb2,
        qbuf, logits, segmax, nT32, E);

    // 3) exp + segsum
    long long tot = E * HEADS;
    softmax_edge_kernel<<<(int)((tot + 255) / 256), 256, 0, stream>>>(
        edge_idx, logits, segmax, segsum, E);

    // 4) edge v-MLP + weighted scatter
    edge_v_agg_kernel<<<nT32, 32, 0, stream>>>(
        h, edge_attr, edge_idx, hvW1p, hvb1, hvg, hvbe, hvW2p, hvb2,
        logits, segsum, agg, nT32, E);

    // 5) node_out MLP + residual
    node_out_kernel<<<(nTilesN + 1) / 2, 64, 0, stream>>>(
        h, agg, noW1p, nob1, nog, nobe, noW2p, nob2, (float*)d_out, nTilesN);
}
